// MultiPeakAwareAttention_46188078301771
// MI455X (gfx1250) — compile-verified, matrix operand order checked by reading, not run on hardware
//
#include <hip/hip_runtime.h>
#include <math.h>

#define B_    8
#define H_    48
#define W_    48
#define N_    (H_*W_)      // 2304
#define C_    256
#define NP1   (N_+1)       // 2305
#define K_TOP 8
#define INV_TEMP 20.0f

typedef __attribute__((ext_vector_type(16))) __bf16 v16bf;
typedef __attribute__((ext_vector_type(8)))  float  v8f;

union FragU { v16bf v; uint4 q[2]; };

__device__ __forceinline__ unsigned short f2bf(float x) {
  unsigned int u = __float_as_uint(x);
  unsigned int r = u + 0x7FFFu + ((u >> 16) & 1u);   // round-to-nearest-even
  return (unsigned short)(r >> 16);
}

// ---------------- normalize + convert to bf16 -------------------------------
__global__ void k_normalize(const float* __restrict__ fa, const float* __restrict__ fb,
                            unsigned short* __restrict__ na, unsigned short* __restrict__ nb) {
  int wid = threadIdx.x >> 5, lane = threadIdx.x & 31;
  int row = blockIdx.x * 8 + wid;                 // over 2*B*N rows
  const float* src; unsigned short* dst; int r;
  if (row < B_*N_) { src = fa; dst = na; r = row; }
  else             { src = fb; dst = nb; r = row - B_*N_; }
  src += (size_t)r * C_;
  dst += (size_t)r * C_;
  float x[8]; float ss = 0.f;
  #pragma unroll
  for (int t = 0; t < 8; ++t) { x[t] = src[lane*8 + t]; ss += x[t]*x[t]; }
  #pragma unroll
  for (int off = 16; off; off >>= 1) ss += __shfl_xor(ss, off, 32);
  float inv = 1.0f / fmaxf(sqrtf(ss), 1e-12f);
  unsigned short h[8];
  #pragma unroll
  for (int t = 0; t < 8; ++t) h[t] = f2bf(x[t]*inv);
  uint4 pack;
  pack.x = (unsigned)h[0] | ((unsigned)h[1]<<16);
  pack.y = (unsigned)h[2] | ((unsigned)h[3]<<16);
  pack.z = (unsigned)h[4] | ((unsigned)h[5]<<16);
  pack.w = (unsigned)h[6] | ((unsigned)h[7]<<16);
  *(uint4*)(dst + lane*8) = pack;
}

// ---------------- WMMA GEMM: raw_sim = nA * nB^T ----------------------------
__device__ __forceinline__ v16bf load_frag(const unsigned short* rowp, int k0, int hi) {
  FragU f;
  const unsigned short* p = rowp + k0 + hi*8;     // K runs: [hi*8, +8) and [16+hi*8, +8)
  f.q[0] = *(const uint4*)p;
  f.q[1] = *(const uint4*)(p + 16);
  return f.v;
}

__global__ void k_gemm_sim(const unsigned short* __restrict__ na,
                           const unsigned short* __restrict__ nb,
                           float* __restrict__ sim) {
  const int MB = N_/64, NBt = N_/128;             // 36 x 18 block tiles of 64x128
  int blk = blockIdx.x;
  int b   = blk / (MB*NBt);
  int rem = blk % (MB*NBt);
  int mb = rem / NBt, nbl = rem % NBt;
  int wid = threadIdx.x >> 5, lane = threadIdx.x & 31;
  int wm = wid >> 2, wn = wid & 3;                // 2x4 waves, each 32x32
  int lo = lane & 15, hi = lane >> 4;
  int m0 = mb*64  + wm*32;
  int n0 = nbl*128 + wn*32;
  const unsigned short* A  = na + (size_t)b*N_*C_;
  const unsigned short* Bp = nb + (size_t)b*N_*C_;
  const unsigned short* ar0 = A  + (size_t)(m0 + lo)*C_;
  const unsigned short* ar1 = A  + (size_t)(m0 + 16 + lo)*C_;
  const unsigned short* br0 = Bp + (size_t)(n0 + lo)*C_;
  const unsigned short* br1 = Bp + (size_t)(n0 + 16 + lo)*C_;
  v8f acc00 = {}, acc01 = {}, acc10 = {}, acc11 = {};
  #pragma unroll
  for (int k0 = 0; k0 < C_; k0 += 32) {
    v16bf a0 = load_frag(ar0, k0, hi);
    v16bf a1 = load_frag(ar1, k0, hi);
    v16bf b0 = load_frag(br0, k0, hi);
    v16bf b1 = load_frag(br1, k0, hi);
    acc00 = __builtin_amdgcn_wmma_f32_16x16x32_bf16(false, a0, false, b0, (short)0, acc00, false, false);
    acc01 = __builtin_amdgcn_wmma_f32_16x16x32_bf16(false, a0, false, b1, (short)0, acc01, false, false);
    acc10 = __builtin_amdgcn_wmma_f32_16x16x32_bf16(false, a1, false, b0, (short)0, acc10, false, false);
    acc11 = __builtin_amdgcn_wmma_f32_16x16x32_bf16(false, a1, false, b1, (short)0, acc11, false, false);
  }
  float* Cp = sim + (size_t)b*N_*N_;
  #pragma unroll
  for (int r = 0; r < 8; ++r) {
    int mr0 = m0 + r + 8*hi;
    int mr1 = mr0 + 16;
    Cp[(size_t)mr0*N_ + n0 + lo]      = acc00[r];
    Cp[(size_t)mr0*N_ + n0 + 16 + lo] = acc01[r];
    Cp[(size_t)mr1*N_ + n0 + lo]      = acc10[r];
    Cp[(size_t)mr1*N_ + n0 + 16 + lo] = acc11[r];
  }
}

// ---------------- Sinkhorn ---------------------------------------------------
__global__ void k_zero(float* p, int n) {
  int i = blockIdx.x*256 + threadIdx.x;
  if (i < n) p[i] = 0.f;
}

// u update: one wave per row i (incl. dustbin row i==N_), online logsumexp
__global__ void k_sinkhorn_u(const float* __restrict__ sim,
                             const float* __restrict__ v,
                             float* __restrict__ u) {
  int wid = threadIdx.x >> 5, lane = threadIdx.x & 31;
  int row = blockIdx.x * 8 + wid;                 // 0 .. B*NP1-1 (exact)
  int b = row / NP1, i = row % NP1;
  const float* rp = sim + ((size_t)b*N_ + (i < N_ ? i : 0))*N_;
  const float* vp = v + b*NP1;
  float m = -INFINITY, s = 0.f;
  for (int j = lane; j < NP1; j += 32) {
    float z = (i < N_ && j < N_) ? rp[j]*INV_TEMP : 0.f;
    float val = z + vp[j];
    if (val > m) { s = s*expf(m - val) + 1.f; m = val; }
    else         { s += expf(val - m); }
  }
  #pragma unroll
  for (int off = 16; off; off >>= 1) {
    float mo = __shfl_xor(m, off, 32);
    float so = __shfl_xor(s, off, 32);
    float nm = fmaxf(m, mo);
    s = s*expf(m - nm) + so*expf(mo - nm);
    m = nm;
  }
  if (lane == 0) {
    const float norm = -logf((float)(2*N_));
    float lmu = (i < N_) ? norm : (logf((float)N_) + norm);
    u[b*NP1 + i] = lmu - (m + logf(s));
  }
}

// v update: one thread per column j; row-major reads are coalesced across lanes
__global__ void k_sinkhorn_v(const float* __restrict__ sim,
                             const float* __restrict__ u,
                             float* __restrict__ v) {
  __shared__ float su[NP1];
  const int bpb = (NP1 + 255) / 256;              // 10 blocks per batch
  int b = blockIdx.x / bpb;
  int j = (blockIdx.x % bpb) * 256 + threadIdx.x;
  for (int i = threadIdx.x; i < NP1; i += 256) su[i] = u[b*NP1 + i];
  __syncthreads();
  if (j >= NP1) return;
  const float* sp = sim + (size_t)b*N_*N_;
  float m = -INFINITY, s = 0.f;
  for (int i = 0; i < N_; ++i) {
    float z = (j < N_) ? sp[(size_t)i*N_ + j]*INV_TEMP : 0.f;
    float val = z + su[i];
    if (val > m) { s = s*expf(m - val) + 1.f; m = val; }
    else         { s += expf(val - m); }
  }
  { // dustbin row i==N_: z = 0
    float val = su[N_];
    if (val > m) { s = s*expf(m - val) + 1.f; m = val; }
    else         { s += expf(val - m); }
  }
  const float norm = -logf((float)(2*N_));
  float lnu = (j < N_) ? norm : (logf((float)N_) + norm);
  v[b*NP1 + j] = lnu - (m + logf(s));
}

// ---------------- exp + row-sum + entropy + top-8 ---------------------------
__global__ void k_topk(const float* __restrict__ sim,
                       const float* __restrict__ u, const float* __restrict__ v,
                       const float* __restrict__ posA, const float* __restrict__ posB,
                       float* __restrict__ out_ent,
                       float* __restrict__ tv, float* __restrict__ tpos,
                       float* __restrict__ aplane, float* __restrict__ qplane) {
  __shared__ float sp[N_];
  __shared__ float rv[256];
  __shared__ int   ri[256];
  __shared__ float wval[K_TOP];
  __shared__ int   widx[K_TOP];
  __shared__ float sS;
  int b = blockIdx.x / N_, i = blockIdx.x % N_;
  int tid = threadIdx.x;
  float ui = u[b*NP1 + i];
  const float* sr = sim + ((size_t)b*N_ + i)*N_;
  const float* vp = v + b*NP1;
  float ls = 0.f;
  for (int j = tid; j < N_; j += 256) {
    float p = expf(sr[j]*INV_TEMP + ui + vp[j]);
    sp[j] = p; ls += p;
  }
  rv[tid] = ls; __syncthreads();
  for (int sft = 128; sft; sft >>= 1) { if (tid < sft) rv[tid] += rv[tid+sft]; __syncthreads(); }
  if (tid == 0) sS = rv[0] + 1e-8f;
  __syncthreads();
  float S = sS;
  float le = 0.f;
  for (int j = tid; j < N_; j += 256) {
    float pd = sp[j] / S;
    le += pd * logf(pd + 1e-8f);
  }
  rv[tid] = le; __syncthreads();
  for (int sft = 128; sft; sft >>= 1) { if (tid < sft) rv[tid] += rv[tid+sft]; __syncthreads(); }
  if (tid == 0) out_ent[b*N_ + i] = -rv[0];
  __syncthreads();
  for (int k = 0; k < K_TOP; ++k) {
    float bv = -1.f; int bi = 0;
    for (int j = tid; j < N_; j += 256) {
      float pv = sp[j];
      if (pv > bv) { bv = pv; bi = j; }
    }
    rv[tid] = bv; ri[tid] = bi; __syncthreads();
    for (int sft = 128; sft; sft >>= 1) {
      if (tid < sft) {
        float v2 = rv[tid+sft]; int i2 = ri[tid+sft];
        if (v2 > rv[tid] || (v2 == rv[tid] && i2 < ri[tid])) { rv[tid] = v2; ri[tid] = i2; }
      }
      __syncthreads();
    }
    if (tid == 0) { wval[k] = rv[0]; widx[k] = ri[0]; sp[ri[0]] = -1.f; }
    __syncthreads();
  }
  if (tid < K_TOP) {
    int k = tid; int idx = widx[k];
    float px = posB[((size_t)b*N_ + idx)*2 + 0];
    float py = posB[((size_t)b*N_ + idx)*2 + 1];
    size_t base = ((size_t)b*N_ + i)*K_TOP + k;
    tv[base] = wval[k];
    tpos[base*2+0] = px; tpos[base*2+1] = py;
    float ax = posA[((size_t)b*N_ + i)*2 + 0];
    float ay = posA[((size_t)b*N_ + i)*2 + 1];
    float dx = px - ax, dy = py - ay;
    aplane[(size_t)(b*K_TOP + k)*N_ + i] = dx + dy;          // sum over both channels
    qplane[(size_t)(b*K_TOP + k)*N_ + i] = dx*dx + dy*dy;
  }
}

// ---------------- geo: 7x7 box-sum variance over 48x48 ----------------------
__global__ void k_geo(const float* __restrict__ aplane, const float* __restrict__ qplane,
                      float* __restrict__ geo) {
  __shared__ float sa[N_];
  __shared__ float sq[N_];
  int bk = blockIdx.x;                            // b*8 + k
  int tid = threadIdx.x;
  const float* ap = aplane + (size_t)bk * N_;
  const float* qp = qplane + (size_t)bk * N_;
  for (int p = tid; p < N_; p += 256) { sa[p] = ap[p]; sq[p] = qp[p]; }
  __syncthreads();
  int b = bk / K_TOP, k = bk % K_TOP;
  const float nwin = 98.f;                        // 2 channels * 7*7, zero-padded
  for (int p = tid; p < N_; p += 256) {
    int y = p / W_, x = p % W_;
    float s1 = 0.f, s2 = 0.f;
    for (int dy = -3; dy <= 3; ++dy) {
      int yy = y + dy; if (yy < 0 || yy >= H_) continue;
      int rowb = yy * W_;
      for (int dx = -3; dx <= 3; ++dx) {
        int xx = x + dx; if (xx < 0 || xx >= W_) continue;
        s1 += sa[rowb + xx]; s2 += sq[rowb + xx];
      }
    }
    float var = (s2 - s1*s1/nwin) / (nwin - 1.f);
    geo[((size_t)b*N_ + p)*K_TOP + k] = 1.0f / (1.0f + var*100.f);
  }
}

// ---------------- softmax over K, weighted warp -----------------------------
__global__ void k_final(const float* __restrict__ tv, const float* __restrict__ geo,
                        const float* __restrict__ tpos, float* __restrict__ warp) {
  int idx = blockIdx.x*256 + threadIdx.x;         // b*N + i  (exact grid)
  float c[K_TOP];
  float m = -INFINITY;
  #pragma unroll
  for (int k = 0; k < K_TOP; ++k) {
    c[k] = (tv[(size_t)idx*K_TOP + k] + 1.5f*geo[(size_t)idx*K_TOP + k]) * INV_TEMP;
    m = fmaxf(m, c[k]);
  }
  float sum = 0.f, wx = 0.f, wy = 0.f;
  #pragma unroll
  for (int k = 0; k < K_TOP; ++k) {
    float w = expf(c[k] - m);
    sum += w;
    wx += w * tpos[((size_t)idx*K_TOP + k)*2 + 0];
    wy += w * tpos[((size_t)idx*K_TOP + k)*2 + 1];
  }
  warp[idx*2+0] = wx/sum;
  warp[idx*2+1] = wy/sum;
}

// ---------------- host side --------------------------------------------------
extern "C" void kernel_launch(void* const* d_in, const int* in_sizes, int n_in,
                              void* d_out, int out_size, void* d_ws, size_t ws_size,
                              hipStream_t stream) {
  (void)in_sizes; (void)n_in; (void)out_size; (void)ws_size;
  const float* fa = (const float*)d_in[0];
  const float* fb = (const float*)d_in[1];
  const float* pa = (const float*)d_in[2];
  const float* pb = (const float*)d_in[3];
  // d_in[4], d_in[5] are H,W device scalars; fixed at 48 per setup_inputs.

  float* out      = (float*)d_out;
  float* out_warp = out;                                   // [B,N,2]
  float* out_ent  = out + (size_t)B_*N_*2;                 // [B,N]
  float* out_sim  = out + (size_t)B_*N_*2 + (size_t)B_*N_; // [B,N,N]

  char* ws = (char*)d_ws;
  size_t off = 0;
  auto take = [&](size_t bytes) -> char* {
    char* p = ws + off;
    off = (off + bytes + 255) & ~(size_t)255;
    return p;
  };
  unsigned short* na = (unsigned short*)take((size_t)B_*N_*C_*2);
  unsigned short* nb = (unsigned short*)take((size_t)B_*N_*C_*2);
  float* u      = (float*)take((size_t)B_*NP1*4);
  float* v      = (float*)take((size_t)B_*NP1*4);
  float* tv     = (float*)take((size_t)B_*N_*K_TOP*4);
  float* tpos   = (float*)take((size_t)B_*N_*K_TOP*2*4);
  float* aplane = (float*)take((size_t)B_*K_TOP*N_*4);
  float* qplane = (float*)take((size_t)B_*K_TOP*N_*4);
  float* geo    = (float*)take((size_t)B_*K_TOP*N_*4);

  k_normalize<<<2*B_*N_/8, 256, 0, stream>>>(fa, fb, na, nb);
  k_gemm_sim<<<B_*(N_/64)*(N_/128), 256, 0, stream>>>(na, nb, out_sim);

  int nz = B_*NP1;
  k_zero<<<(nz+255)/256, 256, 0, stream>>>(u, nz);
  k_zero<<<(nz+255)/256, 256, 0, stream>>>(v, nz);
  const int bpb = (NP1 + 255) / 256;
  for (int it = 0; it < 5; ++it) {
    k_sinkhorn_u<<<B_*NP1/8, 256, 0, stream>>>(out_sim, v, u);
    k_sinkhorn_v<<<B_*bpb, 256, 0, stream>>>(out_sim, u, v);
  }

  k_topk<<<B_*N_, 256, 0, stream>>>(out_sim, u, v, pa, pb, out_ent,
                                    tv, tpos, aplane, qplane);
  k_geo<<<B_*K_TOP, 256, 0, stream>>>(aplane, qplane, geo);
  k_final<<<B_*N_/256, 256, 0, stream>>>(tv, geo, tpos, out_warp);
}
